// MeasureDistance_5403068858565
// MI455X (gfx1250) — matrix operand order repaired and checked by reading, not compiled, not run on hardware
//
#include <hip/hip_runtime.h>
#include <hip/hip_bf16.h>
#include <math.h>

typedef unsigned short u16;
typedef unsigned int   u32;
typedef __attribute__((ext_vector_type(8)))  float  v8f;
typedef __attribute__((ext_vector_type(8)))  __bf16 v8bf;
typedef __attribute__((ext_vector_type(16))) __bf16 v16bf;

#define Bn 4
#define Nn 4096
#define Dn 256
#define LDK 40   // u16 stride of LDS tiles (32 + 8 pad, keeps 16B alignment)
#define NTILE 32 // 4096/128 tiles per dimension for the fused dual-LSE pass

// ---------------------------------------------------------------- helpers
static __device__ __forceinline__ u16 f32_to_bf16(float f) {
  u32 u = __float_as_uint(f);
  u32 r = u + 0x7FFFu + ((u >> 16) & 1u);   // round-to-nearest-even
  return (u16)(r >> 16);
}
static __device__ __forceinline__ float bfbits_to_f(u16 h) {
  return __uint_as_float(((u32)h) << 16);
}
static __device__ __forceinline__ v16bf cat8(v8bf a, v8bf b) {
  return __builtin_shufflevector(a, b, 0,1,2,3,4,5,6,7,8,9,10,11,12,13,14,15);
}
// online logsumexp state update: (m,a) <- (m,a) combined with value v
static __device__ __forceinline__ void lse_push(float& m, float& a, float v) {
  float nm = fmaxf(m, v);
  a = a * __expf(m - nm) + __expf(v - nm);
  m = nm;
}
// merge two (m,a) states
static __device__ __forceinline__ void lse_merge(float& m, float& a, float om, float oa) {
  float nm = fmaxf(m, om);
  a = a * __expf(m - nm) + oa * __expf(om - nm);
  m = nm;
}

// ------------------------------------------------- norms + bf16 hi/lo split
__launch_bounds__(256)
__global__ void prep_kernel(const float* __restrict__ src, u16* __restrict__ hi,
                            u16* __restrict__ lo, float* __restrict__ nrm) {
  int lane = threadIdx.x & 31;
  int wave = threadIdx.x >> 5;
  int r = blockIdx.x * 8 + wave;                  // one wave per row
  const float* p = src + (size_t)r * Dn;
  int c = lane * 8;
  float4 v0 = *(const float4*)(p + c);
  float4 v1 = *(const float4*)(p + c + 4);
  float vv[8] = {v0.x, v0.y, v0.z, v0.w, v1.x, v1.y, v1.z, v1.w};
  float ss = 0.f;
  u16 hb[8], lb[8];
#pragma unroll
  for (int j = 0; j < 8; ++j) {
    float f = vv[j];
    ss += f * f;
    u16 h = f32_to_bf16(f);
    hb[j] = h;
    lb[j] = f32_to_bf16(f - bfbits_to_f(h));
  }
  uint4 hp, lp;
  hp.x = (u32)hb[0] | ((u32)hb[1] << 16);
  hp.y = (u32)hb[2] | ((u32)hb[3] << 16);
  hp.z = (u32)hb[4] | ((u32)hb[5] << 16);
  hp.w = (u32)hb[6] | ((u32)hb[7] << 16);
  lp.x = (u32)lb[0] | ((u32)lb[1] << 16);
  lp.y = (u32)lb[2] | ((u32)lb[3] << 16);
  lp.z = (u32)lb[4] | ((u32)lb[5] << 16);
  lp.w = (u32)lb[6] | ((u32)lb[7] << 16);
  *(uint4*)(hi + (size_t)r * Dn + c) = hp;
  *(uint4*)(lo + (size_t)r * Dn + c) = lp;
#pragma unroll
  for (int off = 16; off > 0; off >>= 1) ss += __shfl_xor(ss, off, 32);
  if (lane == 0) nrm[r] = ss;
}

__global__ void logw_kernel(const float* __restrict__ w, float* __restrict__ lw, int n) {
  int i = blockIdx.x * blockDim.x + threadIdx.x;
  if (i < n) lw[i] = logf(w[i]);
}

__global__ void zero_kernel(float* __restrict__ p, int n) {
  for (int i = blockIdx.x * blockDim.x + threadIdx.x; i < n; i += gridDim.x * blockDim.x)
    p[i] = 0.f;
}

// --------------------------------------------------------- WMMA GEMM kernel
// Computes Kout[r,c] = -max(nA[r] + nB[c] - 2 * A_row_r . B_row_c , 0)
// via split-bf16 (hi+lo) 3-WMMA products: hi*hi + hi*lo + lo*hi.
__launch_bounds__(256)
__global__ void gemm_kernel(const u16* __restrict__ Ahi, const u16* __restrict__ Alo,
                            const u16* __restrict__ Bhi, const u16* __restrict__ Blo,
                            const float* __restrict__ nA, const float* __restrict__ nB,
                            float* __restrict__ Kout) {
  __shared__ u16 sAh[128 * LDK], sAl[128 * LDK], sBh[128 * LDK], sBl[128 * LDK];

  const int bb = blockIdx.z;
  const int rb = blockIdx.y * 128, cb = blockIdx.x * 128;
  const int tid = threadIdx.x, lane = tid & 31, wave = tid >> 5;
  const int waveRow = (wave >> 2) * 64;     // 2x4 wave grid: 64(M) x 32(N) each
  const int waveCol = (wave & 3) * 32;

  const u16* Ah = Ahi + ((size_t)bb * Nn + rb) * Dn;
  const u16* Al = Alo + ((size_t)bb * Nn + rb) * Dn;
  const u16* Bh = Bhi + ((size_t)bb * Nn + cb) * Dn;
  const u16* Bl = Blo + ((size_t)bb * Nn + cb) * Dn;

  v8f acc[4][2];
#pragma unroll
  for (int mt = 0; mt < 4; ++mt)
#pragma unroll
    for (int nt = 0; nt < 2; ++nt)
#pragma unroll
      for (int i = 0; i < 8; ++i) acc[mt][nt][i] = 0.f;

  for (int k0 = 0; k0 < Dn; k0 += 32) {
    // ---- stage 128x32 hi/lo tiles for A and B (8-u16 chunks per thread)
#pragma unroll
    for (int j = 0; j < 2; ++j) {
      int chunk = tid * 2 + j;              // 512 chunks cover 128x32
      int row = chunk >> 2;
      int kc = (chunk & 3) * 8;
      size_t g = (size_t)row * Dn + k0 + kc;
      int s = row * LDK + kc;
      *(uint4*)&sAh[s] = *(const uint4*)(Ah + g);
      *(uint4*)&sAl[s] = *(const uint4*)(Al + g);
      *(uint4*)&sBh[s] = *(const uint4*)(Bh + g);
      *(uint4*)&sBl[s] = *(const uint4*)(Bl + g);
    }
    __syncthreads();

    // ---- fragment loads per ISA 16-bit A / B VGPR layouts (wave32)
    const int mrow = lane & 15;
    const int hiHalf = lane >> 4;
    const int kbA = hiHalf * 8;    // A: lanes<16 -> K 0..7 & 16..23 ; lanes>=16 -> +8
    const int kbB = hiHalf * 8;    // B: lanes<16 -> K 0..7 & 16..23 ; lanes>=16 -> +8
    v16bf afh[4], afl[4], bfh[2], bfl[2];
#pragma unroll
    for (int mt = 0; mt < 4; ++mt) {
      int r = (waveRow + mt * 16 + mrow) * LDK;
      afh[mt] = cat8(*(const v8bf*)&sAh[r + kbA], *(const v8bf*)&sAh[r + kbA + 16]);
      afl[mt] = cat8(*(const v8bf*)&sAl[r + kbA], *(const v8bf*)&sAl[r + kbA + 16]);
    }
#pragma unroll
    for (int nt = 0; nt < 2; ++nt) {
      int c = (waveCol + nt * 16 + mrow) * LDK;
      bfh[nt] = cat8(*(const v8bf*)&sBh[c + kbB], *(const v8bf*)&sBh[c + kbB + 16]);
      bfl[nt] = cat8(*(const v8bf*)&sBl[c + kbB], *(const v8bf*)&sBl[c + kbB + 16]);
    }

#pragma unroll
    for (int mt = 0; mt < 4; ++mt)
#pragma unroll
      for (int nt = 0; nt < 2; ++nt) {
        v8f c = acc[mt][nt];
        c = __builtin_amdgcn_wmma_f32_16x16x32_bf16(false, afh[mt], false, bfh[nt],
                                                    (short)0, c, false, false);
        c = __builtin_amdgcn_wmma_f32_16x16x32_bf16(false, afh[mt], false, bfl[nt],
                                                    (short)0, c, false, false);
        c = __builtin_amdgcn_wmma_f32_16x16x32_bf16(false, afl[mt], false, bfh[nt],
                                                    (short)0, c, false, false);
        acc[mt][nt] = c;
      }
    __syncthreads();
  }

  // ---- epilogue: 16x16 f32 C layout -> K = -max(nA + nB - 2*dot, 0)
  const float* na = nA + (size_t)bb * Nn;
  const float* nb = nB + (size_t)bb * Nn;
  const int mBase = (lane >> 4) ? 8 : 0;
  const int nIdx = lane & 15;
#pragma unroll
  for (int mt = 0; mt < 4; ++mt)
#pragma unroll
    for (int nt = 0; nt < 2; ++nt)
#pragma unroll
      for (int i = 0; i < 8; ++i) {
        int rg = rb + waveRow + mt * 16 + mBase + i;
        int cg = cb + waveCol + nt * 16 + nIdx;
        float dval = na[rg] + nb[cg] - 2.0f * acc[mt][nt][i];
        Kout[((size_t)bb * Nn + rg) * Nn + cg] = -fmaxf(dval, 0.0f);
      }
}

// ------------------------------------- extrapolate: row-wise online logsumexp
// out = finalMode ? -LSE : 0.5*(prev + (-LSE)),  LSE over cols of K + (pot+lw)
__launch_bounds__(256)
__global__ void lse_kernel(const float* __restrict__ mat, const float* __restrict__ pot,
                           const float* __restrict__ lw, const float* __restrict__ prevPot,
                           float* __restrict__ outPot, int finalMode) {
  __shared__ float s[Nn];
  const int bb = blockIdx.y, tid = threadIdx.x;
  const float* pp = pot + (size_t)bb * Nn;
  const float* lp = lw + (size_t)bb * Nn;
  for (int i = tid; i < Nn; i += 256) s[i] = pp[i] + lp[i];
  __syncthreads();

  const int lane = tid & 31;
  const int r = blockIdx.x * 8 + (tid >> 5);      // one wave per row
  const float* row = mat + ((size_t)bb * Nn + r) * Nn;

  float m = -__builtin_inff();
  float a = 0.f;
  for (int c0 = lane * 4; c0 < Nn; c0 += 128) {
    __builtin_prefetch(row + c0 + 1024, 0, 0);    // global_prefetch_b8
    float4 v = *(const float4*)(row + c0);
    lse_push(m, a, v.x + s[c0]);
    lse_push(m, a, v.y + s[c0 + 1]);
    lse_push(m, a, v.z + s[c0 + 2]);
    lse_push(m, a, v.w + s[c0 + 3]);
  }
#pragma unroll
  for (int off = 16; off > 0; off >>= 1) {
    float om = __shfl_xor(m, off, 32);
    float oa = __shfl_xor(a, off, 32);
    lse_merge(m, a, om, oa);
  }
  if (lane == 0) {
    float n = -(m + __logf(a));
    size_t o = (size_t)bb * Nn + r;
    if (finalMode) outPot[o] = n;
    else           outPot[o] = 0.5f * (prevPot[o] + n);
  }
}

// --------------------- fused dual-direction LSE over Kxy (single HBM pass)
// Each 128x128 tile contributes online-LSE partials (m,a) for:
//   row direction: val = K[r,c] + sCol[c]   (sCol = potCol + lwCol)
//   col direction: val = K[r,c] + sRow[r]   (sRow = potRow + lwRow)
// Partials are merged exactly by combine_kernel (logsumexp is associative).
__launch_bounds__(256)
__global__ void dual_lse_kernel(const float* __restrict__ mat,
                                const float* __restrict__ potRow, const float* __restrict__ lwRow,
                                const float* __restrict__ potCol, const float* __restrict__ lwCol,
                                float* __restrict__ rowPm, float* __restrict__ rowPa,
                                float* __restrict__ colPm, float* __restrict__ colPa) {
  __shared__ float srow[128], scol[128];
  __shared__ float cmLds[8][132], caLds[8][132];
  const int bb = blockIdx.z, rt = blockIdx.y, ct = blockIdx.x;
  const int rb = rt * 128, cb = ct * 128;
  const int tid = threadIdx.x, tx = tid & 31, ty = tid >> 5;

  if (tid < 128) {
    srow[tid] = potRow[(size_t)bb * Nn + rb + tid] + lwRow[(size_t)bb * Nn + rb + tid];
  } else {
    int t = tid - 128;
    scol[t] = potCol[(size_t)bb * Nn + cb + t] + lwCol[(size_t)bb * Nn + cb + t];
  }
  __syncthreads();

  // per-thread column states for cols tx*4 .. tx*4+3
  float cm[4], ca[4], sc[4];
#pragma unroll
  for (int j = 0; j < 4; ++j) {
    cm[j] = -__builtin_inff();
    ca[j] = 0.f;
    sc[j] = scol[tx * 4 + j];
  }

  const float* base = mat + ((size_t)bb * Nn + rb) * Nn + cb + tx * 4;
  float* rowMout = rowPm + ((size_t)ct * Bn + bb) * Nn + rb;
  float* rowAout = rowPa + ((size_t)ct * Bn + bb) * Nn + rb;

  for (int i = 0; i < 16; ++i) {
    int r = ty + i * 8;                       // wave ty handles rows ty, ty+8, ...
    __builtin_prefetch(base + (size_t)(r + 8) * Nn, 0, 0);
    float4 v = *(const float4*)(base + (size_t)r * Nn);
    float sr = srow[r];
    float vals[4] = {v.x, v.y, v.z, v.w};
    float mr = -__builtin_inff(), ar = 0.f;
#pragma unroll
    for (int j = 0; j < 4; ++j) {
      lse_push(mr, ar, vals[j] + sc[j]);      // row-direction (reduce over c)
      lse_push(cm[j], ca[j], vals[j] + sr);   // col-direction (reduce over r)
    }
    // wave-reduce the row chunk over the 32 lanes (lanes cover 128 cols)
#pragma unroll
    for (int off = 16; off > 0; off >>= 1) {
      float om = __shfl_xor(mr, off, 32);
      float oa = __shfl_xor(ar, off, 32);
      lse_merge(mr, ar, om, oa);
    }
    if (tx == 0) { rowMout[r] = mr; rowAout[r] = ar; }
  }

  // merge column states across the 8 waves
#pragma unroll
  for (int j = 0; j < 4; ++j) {
    cmLds[ty][tx * 4 + j] = cm[j];
    caLds[ty][tx * 4 + j] = ca[j];
  }
  __syncthreads();
  if (tid < 128) {
    float M = cmLds[0][tid], A = caLds[0][tid];
#pragma unroll
    for (int t = 1; t < 8; ++t) lse_merge(M, A, cmLds[t][tid], caLds[t][tid]);
    colPm[((size_t)rt * Bn + bb) * Nn + cb + tid] = M;
    colPa[((size_t)rt * Bn + bb) * Nn + cb + tid] = A;
  }
}

// ---------------------- merge NTILE (m,a) partials per element -> potential
__launch_bounds__(256)
__global__ void combine_kernel(const float* __restrict__ pm, const float* __restrict__ pa,
                               const float* __restrict__ prevPot, float* __restrict__ outPot,
                               int finalMode) {
  int idx = blockIdx.x * 256 + threadIdx.x;   // over Bn*Nn
  int bb = idx >> 12;
  int k = idx & (Nn - 1);
  float M = -__builtin_inff(), A = 0.f;
  for (int t = 0; t < NTILE; ++t) {
    size_t o = ((size_t)t * Bn + bb) * Nn + k;
    lse_merge(M, A, pm[o], pa[o]);
  }
  float n = -(M + __logf(A));
  size_t o = (size_t)bb * Nn + k;
  if (finalMode) outPot[o] = n;
  else           outPot[o] = 0.5f * (prevPot[o] + n);
}

// --------------------------------------------------- final weighted reduction
__launch_bounds__(256)
__global__ void final_kernel(const float* __restrict__ ge, const float* __restrict__ f,
                             const float* __restrict__ aw, const float* __restrict__ fe,
                             const float* __restrict__ g, const float* __restrict__ bw,
                             float* __restrict__ out) {
  __shared__ float red[256];
  int tid = threadIdx.x;
  float acc = 0.f;
  for (int i = tid; i < Bn * Nn; i += 256)
    acc += (ge[i] - f[i]) * aw[i] + (fe[i] - g[i]) * bw[i];
  red[tid] = acc;
  __syncthreads();
  for (int off = 128; off > 0; off >>= 1) {
    if (tid < off) red[tid] += red[tid + off];
    __syncthreads();
  }
  if (tid == 0) out[0] = red[0] / (float)Bn;
}

// ---------------------------------------------------------------- launcher
extern "C" void kernel_launch(void* const* d_in, const int* in_sizes, int n_in,
                              void* d_out, int out_size, void* d_ws, size_t ws_size,
                              hipStream_t stream) {
  (void)in_sizes; (void)n_in; (void)out_size; (void)ws_size;
  const float* x  = (const float*)d_in[0];
  const float* aw = (const float*)d_in[1];
  const float* y  = (const float*)d_in[2];
  const float* bw = (const float*)d_in[3];
  float* out = (float*)d_out;

  const size_t N2 = (size_t)Bn * Nn * Nn;   // per-kernel-matrix float count
  const size_t NR = (size_t)Bn * Nn;        // per-vector float count
  const size_t NE = (size_t)Bn * Nn * Dn;   // bf16 element count
  const size_t NP = (size_t)NTILE * Bn * Nn; // partial (m or a) float count

  char* w = (char*)d_ws;
  float* Kxx = (float*)w; w += N2 * sizeof(float);
  float* Kyy = (float*)w; w += N2 * sizeof(float);
  float* Kxy = (float*)w; w += N2 * sizeof(float);
  u16* xhi = (u16*)w; w += NE * sizeof(u16);
  u16* xlo = (u16*)w; w += NE * sizeof(u16);
  u16* yhi = (u16*)w; w += NE * sizeof(u16);
  u16* ylo = (u16*)w; w += NE * sizeof(u16);
  float* nx = (float*)w; w += NR * sizeof(float);
  float* ny = (float*)w; w += NR * sizeof(float);
  float* la = (float*)w; w += NR * sizeof(float);
  float* lb = (float*)w; w += NR * sizeof(float);
  float* pots = (float*)w; w += 8 * NR * sizeof(float);   // 2 sets x {fxx,fyy,fxy,fyx}
  float* fbuf  = (float*)w; w += NR * sizeof(float);
  float* febuf = (float*)w; w += NR * sizeof(float);
  float* gbuf  = (float*)w; w += NR * sizeof(float);
  float* gebuf = (float*)w; w += NR * sizeof(float);
  float* rowPm = (float*)w; w += NP * sizeof(float);
  float* rowPa = (float*)w; w += NP * sizeof(float);
  float* colPm = (float*)w; w += NP * sizeof(float);
  float* colPa = (float*)w; w += NP * sizeof(float);

  auto P = [&](int set, int which) { return pots + ((size_t)set * 4 + which) * NR; };

  // norms + bf16 hi/lo split
  prep_kernel<<<dim3((unsigned)(NR / 8)), 256, 0, stream>>>(x, xhi, xlo, nx);
  prep_kernel<<<dim3((unsigned)(NR / 8)), 256, 0, stream>>>(y, yhi, ylo, ny);
  logw_kernel<<<dim3((unsigned)((NR + 255) / 256)), 256, 0, stream>>>(aw, la, (int)NR);
  logw_kernel<<<dim3((unsigned)((NR + 255) / 256)), 256, 0, stream>>>(bw, lb, (int)NR);
  zero_kernel<<<dim3(64), 256, 0, stream>>>(pots, (int)(8 * NR));

  // kernel matrices via WMMA (Kyx never materialized: handled by dual-LSE)
  dim3 gg(Nn / 128, Nn / 128, Bn);
  gemm_kernel<<<gg, 256, 0, stream>>>(xhi, xlo, xhi, xlo, nx, nx, Kxx);
  gemm_kernel<<<gg, 256, 0, stream>>>(yhi, ylo, yhi, ylo, ny, ny, Kyy);
  gemm_kernel<<<gg, 256, 0, stream>>>(xhi, xlo, yhi, ylo, nx, ny, Kxy);

  // Sinkhorn loop: 10 iterations, ping-pong potentials
  dim3 lg(Nn / 8, Bn);
  dim3 dg(NTILE, NTILE, Bn);
  dim3 cg((unsigned)(NR / 256));
  for (int it = 0; it < 10; ++it) {
    int o = it & 1, nw = 1 - o;
    lse_kernel<<<lg, 256, 0, stream>>>(Kxx, P(o, 0), la, P(o, 0), P(nw, 0), 0);
    lse_kernel<<<lg, 256, 0, stream>>>(Kyy, P(o, 1), lb, P(o, 1), P(nw, 1), 0);
    // single pass over Kxy feeds both n_xy (row dir) and n_yx (col dir)
    dual_lse_kernel<<<dg, 256, 0, stream>>>(Kxy, P(o, 2), la, P(o, 3), lb,
                                            rowPm, rowPa, colPm, colPa);
    combine_kernel<<<cg, 256, 0, stream>>>(rowPm, rowPa, P(o, 2), P(nw, 2), 0); // f_xy
    combine_kernel<<<cg, 256, 0, stream>>>(colPm, colPa, P(o, 3), P(nw, 3), 0); // f_yx
  }

  // final differentiable extrapolations (forward values identical)
  lse_kernel<<<lg, 256, 0, stream>>>(Kxx, P(0, 0), la, P(0, 0), fbuf, 1);   // f
  lse_kernel<<<lg, 256, 0, stream>>>(Kyy, P(0, 1), lb, P(0, 1), gbuf, 1);   // g
  dual_lse_kernel<<<dg, 256, 0, stream>>>(Kxy, P(0, 2), la, P(0, 3), lb,
                                          rowPm, rowPa, colPm, colPa);
  combine_kernel<<<cg, 256, 0, stream>>>(rowPm, rowPa, P(0, 2), gebuf, 1);  // ge
  combine_kernel<<<cg, 256, 0, stream>>>(colPm, colPa, P(0, 3), febuf, 1);  // fe

  final_kernel<<<1, 256, 0, stream>>>(gebuf, fbuf, aw, febuf, gbuf, bw, out);
}
